// MultiScaleAttention_55946243998191
// MI455X (gfx1250) — compile-verified, matrix operand order checked by reading, not run on hardware
//
#include <hip/hip_runtime.h>
#include <hip/hip_bf16.h>
#include <stdint.h>

typedef __bf16 bf16;
typedef __attribute__((ext_vector_type(16))) __bf16 v16bf;
typedef __attribute__((ext_vector_type(8)))  float  v8f;
typedef __attribute__((ext_vector_type(4)))  unsigned int v4u;

#define NH 12
#define HD 64
#define BB 2
#define TT 2
#define HH 28
#define SS 784
#define LL 1568   // TT*SS
#define CC 768    // NH*HD
#define MROWS 3136  // BB*LL

#define WMMA_BF16(a, b, c) \
  __builtin_amdgcn_wmma_f32_16x16x32_bf16(false, (a), false, (b), (short)0, (c), false, false)

// ---------------------------------------------------------------------------
// 16-bit A/B fragment loader for v_wmma_*_16x16x32_bf16 (layouts symmetric):
// lanes 0-15 -> rows 0-15, elems 0..7 = K 0..7, elems 8..15 = K 16..23;
// lanes 16-31 -> rows 0-15, K 8..15 and K 24..31. Rows row-major, K contiguous
// => two 16-byte loads per lane.
// ---------------------------------------------------------------------------
__device__ __forceinline__ v16bf load_frag(const bf16* base, int lane, int ld) {
  int r    = lane & 15;
  int koff = (lane < 16) ? 0 : 8;
  const bf16* p = base + (size_t)r * ld + koff;
  union { v16bf v; v4u u[2]; } un;
  un.u[0] = *(const v4u*)(p);
  un.u[1] = *(const v4u*)(p + 16);
  return un.v;
}

// ---------------------------------------------------------------------------
// Elementwise f32 -> bf16 convert
// ---------------------------------------------------------------------------
__global__ void cvt_bf16_kernel(const float* __restrict__ in, bf16* __restrict__ out, int n) {
  int i = blockIdx.x * blockDim.x + threadIdx.x;
  if (i < n) out[i] = (bf16)in[i];
}

// ---------------------------------------------------------------------------
// QKV GEMM:  qkv[i,j] = sum_c x[i,c] * qkv_w[j,c]   (A[M,K] * B[N,K]^T)
// M=3136, N=2304, K=768. Wave tile 16x64, block = 4 waves.
// Flat loop body (loads first, MMAs second) + unroll-by-2: the scheduler
// overlaps iteration i+1's loads with iteration i's WMMAs without any
// register rotation (which spilled in the previous round).
// ---------------------------------------------------------------------------
__global__ void gemm_qkv_kernel(const bf16* __restrict__ A, const bf16* __restrict__ Bw,
                                float* __restrict__ qr, float* __restrict__ kr,
                                float* __restrict__ vr) {
  const int K = CC;
  int lane = threadIdx.x & 31;
  int wid  = threadIdx.x >> 5;
  int m0 = blockIdx.x * 16;
  int n0 = blockIdx.y * 256 + wid * 64;

  const bf16* Ab  = A + (size_t)m0 * K;
  const bf16* Bb0 = Bw + (size_t)(n0 +  0) * K;
  const bf16* Bb1 = Bw + (size_t)(n0 + 16) * K;
  const bf16* Bb2 = Bw + (size_t)(n0 + 32) * K;
  const bf16* Bb3 = Bw + (size_t)(n0 + 48) * K;

  v8f zero = {0.f,0.f,0.f,0.f,0.f,0.f,0.f,0.f};
  v8f acc0 = zero, acc1 = zero, acc2 = zero, acc3 = zero;

#pragma unroll 2
  for (int kc = 0; kc < K; kc += 32) {
    v16bf a  = load_frag(Ab  + kc, lane, K);
    v16bf b0 = load_frag(Bb0 + kc, lane, K);
    v16bf b1 = load_frag(Bb1 + kc, lane, K);
    v16bf b2 = load_frag(Bb2 + kc, lane, K);
    v16bf b3 = load_frag(Bb3 + kc, lane, K);
    acc0 = WMMA_BF16(a, b0, acc0);
    acc1 = WMMA_BF16(a, b1, acc1);
    acc2 = WMMA_BF16(a, b2, acc2);
    acc3 = WMMA_BF16(a, b3, acc3);
  }

  // C layout: elem r -> row m0 + r + 8*(lane>=16), col n = (lane&15) + nt*16.
  // M-tiles never straddle the batch boundary (1568 % 16 == 0).
  int rowoff = (lane >= 16) ? 8 : 0;
  int nlo = lane & 15;
  int bb = m0 / LL;
  int ll0 = m0 - bb * LL;
  v8f accs[4] = {acc0, acc1, acc2, acc3};
#pragma unroll
  for (int nt = 0; nt < 4; ++nt) {
    int j = n0 + nt * 16 + nlo;
    int split = j / CC;                 // uniform across lanes (16 | 768)
    int rem = j - split * CC;
    int hh = rem >> 6;
    int dd = rem & 63;
    float* dst = (split == 0) ? qr : ((split == 1) ? kr : vr);
    float* base = dst + (size_t)((bb * NH + hh) * LL + ll0 + rowoff) * HD + dd;
#pragma unroll
    for (int r = 0; r < 8; ++r) base[(size_t)r * HD] = accs[nt][r];
  }
}

// ---------------------------------------------------------------------------
// Depthwise 3x3 conv (same pad) over each (28,28) frame + LayerNorm over hd=64.
// One wave per spatial position, 2 channels per lane, wave32 shfl reductions.
// Writes bf16 row-major, optionally f32 (q), optionally bf16 head-transposed
// [B, nh, hd, L] (v -> contiguous keys for the V^T WMMA fragments).
// ---------------------------------------------------------------------------
__global__ void pool_ln_kernel(const float* __restrict__ src, const float* __restrict__ w,
                               const float* __restrict__ g, const float* __restrict__ bta,
                               bf16* __restrict__ dstb, float* __restrict__ dstf,
                               bf16* __restrict__ dstT) {
  int lane = threadIdx.x & 31;
  int wid  = threadIdx.x >> 5;
  int gid = blockIdx.x * (blockDim.x >> 5) + wid;
  const int TOTAL = BB * NH * TT * SS;
  if (gid >= TOTAL) return;

  int frame = gid / SS;                 // (b*NH + h)*TT + t
  int sp = gid - frame * SS;
  int x = sp / HH, y = sp - x * HH;
  const float* fb = src + (size_t)frame * SS * HD;

  int c0 = lane, c1 = lane + 32;
  float a0 = 0.f, a1 = 0.f;
#pragma unroll
  for (int dy = -1; dy <= 1; ++dy) {
    int xx = x + dy;
    if (xx < 0 || xx >= HH) continue;
#pragma unroll
    for (int dx = -1; dx <= 1; ++dx) {
      int yy = y + dx;
      if (yy < 0 || yy >= HH) continue;
      const float* p  = fb + (size_t)(xx * HH + yy) * HD;
      const float* wk = w + ((dy + 1) * 3 + (dx + 1)) * HD;
      a0 += p[c0] * wk[c0];
      a1 += p[c1] * wk[c1];
    }
  }
  float s = a0 + a1, sq = a0 * a0 + a1 * a1;
#pragma unroll
  for (int m = 1; m < 32; m <<= 1) { s += __shfl_xor(s, m); sq += __shfl_xor(sq, m); }
  float mu   = s * (1.f / 64.f);
  float var  = sq * (1.f / 64.f) - mu * mu;
  float rstd = rsqrtf(var + 1e-5f);
  float o0 = (a0 - mu) * rstd * g[c0] + bta[c0];
  float o1 = (a1 - mu) * rstd * g[c1] + bta[c1];

  size_t base = (size_t)gid * HD;
  if (dstb) { dstb[base + c0] = (bf16)o0; dstb[base + c1] = (bf16)o1; }
  if (dstf) { dstf[base + c0] = o0; dstf[base + c1] = o1; }
  if (dstT) {
    int head = frame / TT;
    int t = frame - head * TT;
    int l = t * SS + sp;
    bf16* hb = dstT + (size_t)head * HD * LL;
    hb[(size_t)c0 * LL + l] = (bf16)o0;
    hb[(size_t)c1 * LL + l] = (bf16)o1;
  }
}

// ---------------------------------------------------------------------------
// Decomposed rel-pos terms:
//   relH[pos,k] = sum_c qf[pos,c] * rel_pos_h[x(pos)-k+27, c]
//   relW[pos,k] = sum_c qf[pos,c] * rel_pos_w[y(pos)-k+27, c]
// ---------------------------------------------------------------------------
__global__ void relhw_kernel(const float* __restrict__ qf, const float* __restrict__ rph,
                             const float* __restrict__ rpw, float* __restrict__ relH,
                             float* __restrict__ relW) {
  int gid = blockIdx.x * blockDim.x + threadIdx.x;
  const int TOT = BB * NH * TT * SS * HH;
  if (gid >= TOT) return;
  int pos = gid / HH;
  int k = gid - pos * HH;
  int sp = pos % SS;
  int x = sp / HH, y = sp - x * HH;
  const float* qrow = qf + (size_t)pos * HD;
  const float* rh = rph + (size_t)(x - k + HH - 1) * HD;
  const float* rw = rpw + (size_t)(y - k + HH - 1) * HD;
  float sh = 0.f, sw = 0.f;
#pragma unroll 8
  for (int c = 0; c < HD; ++c) { sh += qrow[c] * rh[c]; sw += qrow[c] * rw[c]; }
  relH[gid] = sh;
  relW[gid] = sw;
}

// ---------------------------------------------------------------------------
// Fused flash attention per head, TRANSPOSED-SCORE form:
//   S^T = K Q^T  (C-layout: row = key, col = query -> per-lane softmax stats)
//   P^T (exp) is bit-identical to the 16-bit B-fragment layout => no transpose.
//   O^T = V^T P^T with V stored [B,nh,hd,L] => contiguous V^T fragments.
// One wave per 16-query tile; keys in chunks of 32; online softmax.
// ---------------------------------------------------------------------------
__global__ void attn_kernel(const bf16* __restrict__ qb, const bf16* __restrict__ kb,
                            const bf16* __restrict__ vtb, const float* __restrict__ qf,
                            const float* __restrict__ relH, const float* __restrict__ relW,
                            bf16* __restrict__ out) {
  int lane = threadIdx.x & 31;
  int wid  = threadIdx.x >> 5;
  int qt = blockIdx.x * 4 + wid;
  if (qt >= LL / 16) return;
  int h = blockIdx.y, b = blockIdx.z;

  size_t headOff = (size_t)(b * NH + h) * LL * HD;
  const bf16* qh  = qb + headOff;
  const bf16* kh  = kb + headOff;
  const bf16* vth = vtb + headOff;        // [HD][LL]
  const float* qfh = qf + headOff;

  int l0 = qt * 16;
  int tq = l0 / SS;
  int sq0 = l0 - tq * SS;
  int nlo = lane & 15;
  int rowoff = (lane >= 16) ? 8 : 0;

  // per-lane query row of the decomposed rel-pos tables
  size_t relBase = ((size_t)(((b * NH + h) * TT + tq) * SS) + sq0 + nlo) * HH;
  const float* relRowH = relH + relBase;
  const float* relRowW = relW + relBase;

  // Q as B-operand fragments (lane -> query col, elems -> hd contiguous)
  v16bf qB0 = load_frag(qh + (size_t)l0 * HD + 0,  lane, HD);
  v16bf qB1 = load_frag(qh + (size_t)l0 * HD + 32, lane, HD);

  float mrun = -3.0e38f, lrun = 0.f;
  v8f zero = {0.f,0.f,0.f,0.f,0.f,0.f,0.f,0.f};
  v8f O0 = zero, O1 = zero, O2 = zero, O3 = zero;

  const float scale = 0.125f;   // hd^-0.5

  for (int kc = 0; kc < LL; kc += 32) {
    // ---- all loads up front (latency hidden behind MMAs + softmax) -----
    v16bf kA0a = load_frag(kh + (size_t)(kc +  0) * HD + 0,  lane, HD);
    v16bf kA0b = load_frag(kh + (size_t)(kc +  0) * HD + 32, lane, HD);
    v16bf kA1a = load_frag(kh + (size_t)(kc + 16) * HD + 0,  lane, HD);
    v16bf kA1b = load_frag(kh + (size_t)(kc + 16) * HD + 32, lane, HD);
    v16bf vA0 = load_frag(vth + (size_t) 0 * LL + kc, lane, LL);
    v16bf vA1 = load_frag(vth + (size_t)16 * LL + kc, lane, LL);
    v16bf vA2 = load_frag(vth + (size_t)32 * LL + kc, lane, LL);
    v16bf vA3 = load_frag(vth + (size_t)48 * LL + kc, lane, LL);

    // ---- S^T tiles: rows = keys, cols = queries ------------------------
    v8f s0 = zero, s1 = zero;
    s0 = WMMA_BF16(kA0a, qB0, s0);
    s0 = WMMA_BF16(kA0b, qB1, s0);
    s1 = WMMA_BF16(kA1a, qB0, s1);
    s1 = WMMA_BF16(kA1b, qB1, s1);

    // ---- scale + diagonal-temporal-block rel bias ----------------------
    // element r of subtile: key = keybase + r + rowoff, query = l0 + nlo
#pragma unroll
    for (int st = 0; st < 2; ++st) {
      v8f& c = (st == 0) ? s0 : s1;
      int keybase = kc + st * 16;       // 16-key subtile never straddles t
      int tk = keybase / SS;
      if (tk == tq) {
        int ksp = keybase + rowoff - tk * SS;
        int kx = ksp / HH;
        int ky = ksp - kx * HH;
#pragma unroll
        for (int r = 0; r < 8; ++r) {
          c[r] = c[r] * scale + relRowH[kx] + relRowW[ky];
          if (++ky == HH) { ky = 0; ++kx; }
        }
      } else {
#pragma unroll
        for (int r = 0; r < 8; ++r) c[r] *= scale;
      }
    }

    // ---- online softmax: stats are per-lane (query = lane&15) ----------
    float m_loc = s0[0];
#pragma unroll
    for (int r = 1; r < 8; ++r) m_loc = fmaxf(m_loc, s0[r]);
#pragma unroll
    for (int r = 0; r < 8; ++r) m_loc = fmaxf(m_loc, s1[r]);
    m_loc = fmaxf(m_loc, __shfl_xor(m_loc, 16));
    float nm = fmaxf(mrun, m_loc);
    float alpha = __expf(mrun - nm);
    mrun = nm;

    // P^T fragment: elems 0..7 <- subtile0 rows, elems 8..15 <- subtile1 rows
    v16bf pf;
    float ps = 0.f;
#pragma unroll
    for (int r = 0; r < 8; ++r) {
      float e0 = __expf(s0[r] - nm);
      float e1 = __expf(s1[r] - nm);
      ps += e0 + e1;
      pf[r]     = (bf16)e0;
      pf[r + 8] = (bf16)e1;
    }
    ps += __shfl_xor(ps, 16);
    lrun = lrun * alpha + ps;

    O0 *= alpha; O1 *= alpha; O2 *= alpha; O3 *= alpha;

    // ---- O^T += V^T * P^T ---------------------------------------------
    O0 = WMMA_BF16(vA0, pf, O0);
    O1 = WMMA_BF16(vA1, pf, O1);
    O2 = WMMA_BF16(vA2, pf, O2);
    O3 = WMMA_BF16(vA3, pf, O3);
  }

  // ---- epilogue: normalize, + q residual, write [B, L, C] bf16 ---------
  float inv = 1.f / lrun;
  int l = l0 + nlo;
  const float* qrow = qfh + (size_t)l * HD;
  bf16* orow = out + (size_t)(b * LL + l) * CC + h * HD;
  v8f Os[4] = {O0, O1, O2, O3};
#pragma unroll
  for (int nt = 0; nt < 4; ++nt) {
    int d0 = nt * 16 + rowoff;
    union { v4u u; bf16 e[8]; } pk;
#pragma unroll
    for (int r = 0; r < 8; ++r)
      pk.e[r] = (bf16)(Os[nt][r] * inv + qrow[d0 + r]);
    *(v4u*)(orow + d0) = pk.u;          // 16B aligned (d0 multiple of 8)
  }
}

// ---------------------------------------------------------------------------
// Output projection: Y[i,j] = sum_c A[i,c]*out_w[j,c] + out_b[j]  (f32 out)
// ---------------------------------------------------------------------------
__global__ void gemm_proj_kernel(const bf16* __restrict__ A, const bf16* __restrict__ Bw,
                                 const float* __restrict__ bias, float* __restrict__ out) {
  const int K = CC;
  int lane = threadIdx.x & 31;
  int wid  = threadIdx.x >> 5;
  int m0 = blockIdx.x * 16;
  int n0 = blockIdx.y * 256 + wid * 64;

  const bf16* Ab  = A + (size_t)m0 * K;
  const bf16* Bb0 = Bw + (size_t)(n0 +  0) * K;
  const bf16* Bb1 = Bw + (size_t)(n0 + 16) * K;
  const bf16* Bb2 = Bw + (size_t)(n0 + 32) * K;
  const bf16* Bb3 = Bw + (size_t)(n0 + 48) * K;

  v8f zero = {0.f,0.f,0.f,0.f,0.f,0.f,0.f,0.f};
  v8f acc0 = zero, acc1 = zero, acc2 = zero, acc3 = zero;

#pragma unroll 2
  for (int kc = 0; kc < K; kc += 32) {
    v16bf a  = load_frag(Ab  + kc, lane, K);
    v16bf b0 = load_frag(Bb0 + kc, lane, K);
    v16bf b1 = load_frag(Bb1 + kc, lane, K);
    v16bf b2 = load_frag(Bb2 + kc, lane, K);
    v16bf b3 = load_frag(Bb3 + kc, lane, K);
    acc0 = WMMA_BF16(a, b0, acc0);
    acc1 = WMMA_BF16(a, b1, acc1);
    acc2 = WMMA_BF16(a, b2, acc2);
    acc3 = WMMA_BF16(a, b3, acc3);
  }

  int rowoff = (lane >= 16) ? 8 : 0;
  int nlo = lane & 15;
  v8f accs[4] = {acc0, acc1, acc2, acc3};
#pragma unroll
  for (int nt = 0; nt < 4; ++nt) {
    int j = n0 + nt * 16 + nlo;
    float bj = bias[j];
#pragma unroll
    for (int r = 0; r < 8; ++r)
      out[(size_t)(m0 + r + rowoff) * CC + j] = accs[nt][r] + bj;
  }
}

// ---------------------------------------------------------------------------
// Host-side launcher
// ---------------------------------------------------------------------------
extern "C" void kernel_launch(void* const* d_in, const int* in_sizes, int n_in,
                              void* d_out, int out_size, void* d_ws, size_t ws_size,
                              hipStream_t stream) {
  (void)in_sizes; (void)n_in; (void)out_size; (void)ws_size;

  const float* x        = (const float*)d_in[0];
  const float* qkv_w    = (const float*)d_in[1];
  const float* out_w    = (const float*)d_in[2];
  const float* out_b    = (const float*)d_in[3];
  const float* q_pool_w = (const float*)d_in[4];
  const float* k_pool_w = (const float*)d_in[5];
  const float* v_pool_w = (const float*)d_in[6];
  const float* q_ln_g   = (const float*)d_in[7];
  const float* q_ln_b   = (const float*)d_in[8];
  const float* k_ln_g   = (const float*)d_in[9];
  const float* k_ln_b   = (const float*)d_in[10];
  const float* v_ln_g   = (const float*)d_in[11];
  const float* v_ln_b   = (const float*)d_in[12];
  const float* rph      = (const float*)d_in[13];
  const float* rpw      = (const float*)d_in[14];
  float* Y = (float*)d_out;

  // ---- workspace carve-up (256B aligned) ----
  uint8_t* ws = (uint8_t*)d_ws;
  size_t off = 0;
  auto carve = [&](size_t bytes) -> void* {
    void* p = ws + off;
    off += (bytes + 255) & ~(size_t)255;
    return p;
  };
  const size_t NX   = (size_t)MROWS * CC;         // 2,408,448
  const size_t NQKV = (size_t)3 * CC * CC;
  const size_t NOW  = (size_t)CC * CC;
  const size_t NREL = (size_t)BB * NH * TT * SS * HH;

  bf16*  xb     = (bf16*)carve(NX * 2);
  bf16*  wqkvb  = (bf16*)carve(NQKV * 2);
  bf16*  wob    = (bf16*)carve(NOW * 2);
  float* q_raw  = (float*)carve(NX * 4);
  float* k_raw  = (float*)carve(NX * 4);
  float* v_raw  = (float*)carve(NX * 4);
  bf16*  qbuf   = (bf16*)carve(NX * 2);
  bf16*  kbuf   = (bf16*)carve(NX * 2);
  bf16*  vtbuf  = (bf16*)carve(NX * 2);           // head-transposed [B,nh,hd,L]
  float* qf     = (float*)carve(NX * 4);
  float* relH   = (float*)carve(NREL * 4);
  float* relW   = (float*)carve(NREL * 4);
  bf16*  attn_o = (bf16*)carve(NX * 2);

  // 1) bf16 converts
  cvt_bf16_kernel<<<(int)((NX + 255) / 256), 256, 0, stream>>>(x, xb, (int)NX);
  cvt_bf16_kernel<<<(int)((NQKV + 255) / 256), 256, 0, stream>>>(qkv_w, wqkvb, (int)NQKV);
  cvt_bf16_kernel<<<(int)((NOW + 255) / 256), 256, 0, stream>>>(out_w, wob, (int)NOW);

  // 2) QKV GEMM (M=3136, N=2304, K=768)
  gemm_qkv_kernel<<<dim3(MROWS / 16, (3 * CC) / 256), 128, 0, stream>>>(
      xb, wqkvb, q_raw, k_raw, v_raw);

  // 3) conv-pool + LayerNorm
  {
    int total = BB * NH * TT * SS;          // 37632 positions
    int blocks = (total + 7) / 8;           // 8 waves per 256-thread block
    pool_ln_kernel<<<blocks, 256, 0, stream>>>(q_raw, q_pool_w, q_ln_g, q_ln_b,
                                               qbuf, qf, nullptr);
    pool_ln_kernel<<<blocks, 256, 0, stream>>>(k_raw, k_pool_w, k_ln_g, k_ln_b,
                                               kbuf, nullptr, nullptr);
    pool_ln_kernel<<<blocks, 256, 0, stream>>>(v_raw, v_pool_w, v_ln_g, v_ln_b,
                                               nullptr, nullptr, vtbuf);
  }

  // 4) decomposed rel-pos terms
  relhw_kernel<<<(int)((NREL + 255) / 256), 256, 0, stream>>>(qf, rph, rpw, relH, relW);

  // 5) fused flash attention (transposed-score form), bf16 out [B, L, C]
  attn_kernel<<<dim3((LL / 16 + 3) / 4, NH, BB), 128, 0, stream>>>(
      qbuf, kbuf, vtbuf, qf, relH, relW, attn_o);

  // 6) output projection (M=3136, N=768, K=768) + bias -> f32 d_out
  gemm_proj_kernel<<<dim3(MROWS / 16, CC / 256), 128, 0, stream>>>(attn_o, wob, out_b, Y);
}
